// RouterModel_75617194213669
// MI455X (gfx1250) — matrix operand order repaired and checked by reading, not compiled, not run on hardware
//
#include <hip/hip_runtime.h>
#include <math.h>

typedef __attribute__((ext_vector_type(16))) __bf16 v16bf;
typedef __attribute__((ext_vector_type(8)))  __bf16 v8bf;
typedef __attribute__((ext_vector_type(8)))  float  v8f;

#define N_NODES 2048
#define N_EDGES 4096
#define DMODEL  128
#define NHEAD   4
#define DHEAD   32
#define NCLS    16

// ---------------------------------------------------------------- helpers

static __device__ __forceinline__ v8f wmma_bf16(v16bf a, v16bf b, v8f c) {
  // v_wmma_f32_16x16x32_bf16: (neg_a, A, neg_b, B, c_mod, C, reuse_a, reuse_b)
  return __builtin_amdgcn_wmma_f32_16x16x32_bf16(false, a, false, b, (short)0, c,
                                                 false, false);
}

static __device__ __forceinline__ float rmax16(float v) {
#pragma unroll
  for (int m = 1; m < 16; m <<= 1) v = fmaxf(v, __shfl_xor(v, m, 32));
  return v;
}
static __device__ __forceinline__ float rsum16(float v) {
#pragma unroll
  for (int m = 1; m < 16; m <<= 1) v += __shfl_xor(v, m, 32);
  return v;
}

// load 8 contiguous bf16 (16B aligned) into dst[at..at+7]; works for LDS & global
static __device__ __forceinline__ void ld8(v16bf& dst, int at, const __bf16* p) {
  v8bf t = *(const v8bf*)p;
#pragma unroll
  for (int i = 0; i < 8; ++i) dst[at + i] = t[i];
}

static __device__ __forceinline__ v16bf pack16(float4 a, float4 b, float4 c, float4 d) {
  v16bf r;
  r[0]=(__bf16)a.x;  r[1]=(__bf16)a.y;  r[2]=(__bf16)a.z;  r[3]=(__bf16)a.w;
  r[4]=(__bf16)b.x;  r[5]=(__bf16)b.y;  r[6]=(__bf16)b.z;  r[7]=(__bf16)b.w;
  r[8]=(__bf16)c.x;  r[9]=(__bf16)c.y;  r[10]=(__bf16)c.z; r[11]=(__bf16)c.w;
  r[12]=(__bf16)d.x; r[13]=(__bf16)d.y; r[14]=(__bf16)d.z; r[15]=(__bf16)d.w;
  return r;
}

static __device__ __forceinline__ unsigned int pack2bf(float lo, float hi) {
  union { __bf16 h[2]; unsigned int u; } pk;
  pk.h[0] = (__bf16)lo; pk.h[1] = (__bf16)hi;
  return pk.u;
}

static __device__ __forceinline__ float gelu_tanh(float x) {
  return 0.5f * x * (1.f + tanhf(0.7978845608028654f * (x + 0.044715f * x * x * x)));
}

// ---------------------------------------------------------------- router

__global__ void router_kernel(const float* __restrict__ x, const float* __restrict__ w,
                              const float* __restrict__ b, float* __restrict__ scores,
                              float* __restrict__ xg, int rows) {
  int r = blockIdx.x * blockDim.x + threadIdx.x;
  if (r >= rows) return;
  const float4* xp = (const float4*)(x + (long)r * DMODEL);
  const float4* wp = (const float4*)w;
  float acc = 0.f;
#pragma unroll
  for (int i = 0; i < DMODEL / 4; ++i) {
    float4 xv = xp[i], wv = wp[i];
    acc += xv.x * wv.x + xv.y * wv.y + xv.z * wv.z + xv.w * wv.w;
  }
  float s = 1.f / (1.f + __expf(-(acc + b[0])));
  scores[r] = s;
  float4* og = (float4*)(xg + (long)r * DMODEL);
#pragma unroll
  for (int i = 0; i < DMODEL / 4; ++i) {
    float4 xv = xp[i];
    xv.x *= s; xv.y *= s; xv.z *= s; xv.w *= s;
    og[i] = xv;
  }
}

// ------------------------------------------------- top-k threshold -> key bias
// single block, 1024 threads: bitonic sort 2048 scores (descending) in LDS

__global__ void topk_bias_kernel(const float* __restrict__ scores,
                                 float* __restrict__ key_bias) {
  __shared__ float s[N_NODES];
  __shared__ float th;
  const int t = threadIdx.x;
  s[t] = scores[t];
  s[t + 1024] = scores[t + 1024];
  __syncthreads();
  for (int k = 2; k <= N_NODES; k <<= 1) {
    for (int j = k >> 1; j > 0; j >>= 1) {
      for (int i = t; i < N_NODES; i += 1024) {
        int ixj = i ^ j;
        if (ixj > i) {
          bool down = ((i & k) == 0);       // descending order
          float a = s[i], b2 = s[ixj];
          if (down ? (a < b2) : (a > b2)) { s[i] = b2; s[ixj] = a; }
        }
      }
      __syncthreads();
    }
  }
  if (t == 0) th = s[N_NODES / 2 - 1];      // k = 0.5*N -> index 1023
  __syncthreads();
  float thr = th;
  for (int i = t; i < N_NODES; i += 1024)
    key_bias[i] = (scores[i] >= thr) ? 0.f : -1e9f;
}

// ------------------------------------------------- generic WMMA GEMM
// acc = X[M,K] @ W[K,N] + bias[N]
// outmode 0: f32 row-major, + optional residual / GELU
// outmode 1: bf16 row-major  [M][N]
// outmode 2: bf16 transposed [N][M]  (pairwise-packed b32 stores)
// 1 wave per block; block computes one 16x16 output tile; K in {128,256}

__global__ void gemm_kernel(const float* __restrict__ X, const float* __restrict__ W,
                            const float* __restrict__ bias, const float* __restrict__ res,
                            float* __restrict__ outf, __bf16* __restrict__ outb,
                            int M, int K, int N, int act, int outmode) {
  __shared__ __align__(32) __bf16 wt[16 * 256];   // W^T tile: [n][k], max K = 256
  const int lane = threadIdx.x;
  const int half = lane >> 4, l15 = lane & 15;
  const int gn = blockIdx.y * 16;
  for (int idx = lane; idx < 16 * K; idx += 32) {
    int n = idx / K, kk = idx - n * K;
    wt[idx] = (__bf16)W[(long)kk * N + gn + n];
  }
  __syncthreads();

  v8f acc;
#pragma unroll
  for (int j = 0; j < 8; ++j) acc[j] = 0.f;

  const int row = blockIdx.x * 16 + l15;          // A row for this lane
  for (int kc = 0; kc < K; kc += 32) {
    const float* p0 = X + (long)row * K + kc + half * 8;
    float4 f0 = ((const float4*)p0)[0];
    float4 f1 = ((const float4*)p0)[1];
    const float* p1 = p0 + 16;
    float4 f2 = ((const float4*)p1)[0];
    float4 f3 = ((const float4*)p1)[1];
    v16bf a = pack16(f0, f1, f2, f3);

    v16bf b;
    const __bf16* bp = &wt[l15 * K + kc + half * 8];
    ld8(b, 0, bp);
    ld8(b, 8, bp + 16);

    acc = wmma_bf16(a, b, acc);
  }

  const float bn = bias[gn + l15];
  const int r0 = blockIdx.x * 16 + half * 8;
  if (outmode == 0) {
#pragma unroll
    for (int j = 0; j < 8; ++j) {
      int r = r0 + j;
      float v = acc[j] + bn;
      if (res) v += res[(long)r * N + gn + l15];
      if (act) v = gelu_tanh(v);
      outf[(long)r * N + gn + l15] = v;
    }
  } else if (outmode == 1) {                      // bf16 row-major
#pragma unroll
    for (int j = 0; j < 8; ++j)
      outb[(long)(r0 + j) * N + gn + l15] = (__bf16)(acc[j] + bn);
  } else {                                        // bf16 transposed [N][M]
    unsigned int* op = (unsigned int*)(outb + (long)(gn + l15) * M + r0);
#pragma unroll
    for (int j = 0; j < 8; j += 2)
      op[j >> 1] = pack2bf(acc[j] + bn, acc[j + 1] + bn);
  }
}

// ------------------------------------------------- fused flash attention
// MODE 0: key bias vector (node top-k mask).  MODE 1: line-graph adjacency.
// 1 wave per block; block = (query-tile of 16, head). Br=16, Bc=32, dh=32.
// q,k: bf16 [T][128] row-major; vt: bf16 [128][T] (per-head transposed V).
// B operands are loaded directly from global (L2-resident working set);
// only the P-tile C-layout -> A-layout conversion round-trips through LDS.

template <int MODE>
__global__ void attn_kernel(const __bf16* __restrict__ q, const __bf16* __restrict__ k,
                            const __bf16* __restrict__ vt,
                            const float* __restrict__ key_bias,
                            const int* __restrict__ src, const int* __restrict__ dst,
                            float* __restrict__ out, int T) {
  __shared__ __align__(32) __bf16 Psh[16 * 32];   // [qrow][key]
  const int lane = threadIdx.x;
  const int half = lane >> 4, l15 = lane & 15;
  const int q0 = blockIdx.x * 16;
  const int hoff = blockIdx.y * DHEAD;
  const int khalf = half * 16;
  const float scale = 0.17677669529663687f;       // 1/sqrt(32)

  // A operand: Q tile (16x32) in A-layout (lane row = l15, K chunk by half)
  v16bf aq;
  {
    const __bf16* p = q + (long)(q0 + l15) * DMODEL + hoff + half * 8;
    ld8(aq, 0, p);
    ld8(aq, 8, p + 16);
  }

  int sq[8], dq[8];
  if (MODE == 1) {
#pragma unroll
    for (int j = 0; j < 8; ++j) {
      int r = q0 + half * 8 + j;
      sq[j] = src[r]; dq[j] = dst[r];
    }
  }

  float m_run[8], l_run[8];
  v8f o0, o1;
#pragma unroll
  for (int j = 0; j < 8; ++j) { m_run[j] = -1e30f; l_run[j] = 0.f; o0[j] = 0.f; o1[j] = 0.f; }

  for (int kb = 0; kb < T; kb += 32) {
    // S = Q K^T for 32 keys: two 16x16x32 WMMAs, B straight from global bf16 K
    v16bf b0, b1;
    {
      const __bf16* kp0 = k + (long)(kb + l15) * DMODEL + hoff + half * 8;
      const __bf16* kp1 = k + (long)(kb + 16 + l15) * DMODEL + hoff + half * 8;
      ld8(b0, 0, kp0); ld8(b0, 8, kp0 + 16);
      ld8(b1, 0, kp1); ld8(b1, 8, kp1 + 16);
    }
    v8f zero;
#pragma unroll
    for (int j = 0; j < 8; ++j) zero[j] = 0.f;
    v8f s0 = wmma_bf16(aq, b0, zero);
    v8f s1 = wmma_bf16(aq, b1, zero);

    // scale + mask (C layout: lane holds key column l15, rows half*8+j)
    if (MODE == 0) {
      bool a0 = (key_bias[kb + l15] > -1.f);
      bool a1 = (key_bias[kb + 16 + l15] > -1.f);
#pragma unroll
      for (int j = 0; j < 8; ++j) {
        s0[j] = a0 ? s0[j] * scale : -1e9f;
        s1[j] = a1 ? s1[j] * scale : -1e9f;
      }
    } else {
      int sk0 = src[kb + l15],      dk0 = dst[kb + l15];
      int sk1 = src[kb + 16 + l15], dk1 = dst[kb + 16 + l15];
#pragma unroll
      for (int j = 0; j < 8; ++j) {
        bool a0 = (sq[j]==sk0)||(sq[j]==dk0)||(dq[j]==sk0)||(dq[j]==dk0);
        bool a1 = (sq[j]==sk1)||(sq[j]==dk1)||(dq[j]==sk1)||(dq[j]==dk1);
        s0[j] = a0 ? s0[j] * scale : -1e9f;
        s1[j] = a1 ? s1[j] * scale : -1e9f;
      }
    }

    // online softmax (per-row stats replicated across the 16-lane half)
#pragma unroll
    for (int j = 0; j < 8; ++j) {
      float rm = rmax16(fmaxf(s0[j], s1[j]));
      float nm = fmaxf(m_run[j], rm);
      float alpha = __expf(m_run[j] - nm);
      float p0 = __expf(s0[j] - nm);
      float p1 = __expf(s1[j] - nm);
      s0[j] = p0; s1[j] = p1;
      float rs = rsum16(p0 + p1);
      l_run[j] = l_run[j] * alpha + rs;
      m_run[j] = nm;
      o0[j] *= alpha;
      o1[j] *= alpha;
    }

    // C-layout -> A-layout through LDS (bf16)
    __syncthreads();                              // WAR: prior reads done
#pragma unroll
    for (int j = 0; j < 8; ++j) {
      Psh[(half * 8 + j) * 32 + l15]      = (__bf16)s0[j];
      Psh[(half * 8 + j) * 32 + 16 + l15] = (__bf16)s1[j];
    }
    __syncthreads();

    v16bf ap, bv0, bv1;
    {
      const __bf16* pp = &Psh[l15 * 32 + half * 8];
      ld8(ap, 0, pp);
      ld8(ap, 8, pp + 16);
      // P @ V via transposed-V in global: rows are head dims, keys contiguous
      const __bf16* vb0 = vt + (long)(hoff + l15) * T + kb + half * 8;
      const __bf16* vb1 = vt + (long)(hoff + 16 + l15) * T + kb + half * 8;
      ld8(bv0, 0, vb0); ld8(bv0, 8, vb0 + 16);
      ld8(bv1, 0, vb1); ld8(bv1, 8, vb1 + 16);
    }
    o0 = wmma_bf16(ap, bv0, o0);
    o1 = wmma_bf16(ap, bv1, o1);
  }

#pragma unroll
  for (int j = 0; j < 8; ++j) {
    float inv = 1.f / fmaxf(l_run[j], 1e-30f);
    int row = q0 + half * 8 + j;
    out[(long)row * DMODEL + hoff + l15]      = o0[j] * inv;
    out[(long)row * DMODEL + hoff + 16 + l15] = o1[j] * inv;
  }
}

// ------------------------------------------------- node message gather

__global__ void node_msg_kernel(const float* __restrict__ node_out,
                                const int* __restrict__ src, const int* __restrict__ dst,
                                float* __restrict__ msg) {
  int idx = blockIdx.x * blockDim.x + threadIdx.x;   // E * 32
  int e = idx >> 5, d4 = idx & 31;
  const float4* a = (const float4*)(node_out + (long)src[e] * DMODEL);
  const float4* b = (const float4*)(node_out + (long)dst[e] * DMODEL);
  float4 av = a[d4], bv = b[d4];
  float4 r; r.x = av.x + bv.x; r.y = av.y + bv.y; r.z = av.z + bv.z; r.w = av.w + bv.w;
  ((float4*)(msg + (long)e * DMODEL))[d4] = r;
}

// ---------------------------------------------------------------- launch

extern "C" void kernel_launch(void* const* d_in, const int* in_sizes, int n_in,
                              void* d_out, int out_size, void* d_ws, size_t ws_size,
                              hipStream_t stream) {
  const float* node_f = (const float*)d_in[0];
  const float* edge_f = (const float*)d_in[1];
  const int*   eidx   = (const int*)d_in[2];
  const int*   src    = eidx;
  const int*   dst    = eidx + N_EDGES;
  const float* w_rn = (const float*)d_in[3];  const float* b_rn = (const float*)d_in[4];
  const float* w_re = (const float*)d_in[5];  const float* b_re = (const float*)d_in[6];
  const float* wq_n = (const float*)d_in[7];  const float* bq_n = (const float*)d_in[8];
  const float* wk_n = (const float*)d_in[9];  const float* bk_n = (const float*)d_in[10];
  const float* wv_n = (const float*)d_in[11]; const float* bv_n = (const float*)d_in[12];
  const float* wo_n = (const float*)d_in[13]; const float* bo_n = (const float*)d_in[14];
  const float* wq_e = (const float*)d_in[15]; const float* bq_e = (const float*)d_in[16];
  const float* wk_e = (const float*)d_in[17]; const float* bk_e = (const float*)d_in[18];
  const float* wv_e = (const float*)d_in[19]; const float* bv_e = (const float*)d_in[20];
  const float* wo_e = (const float*)d_in[21]; const float* bo_e = (const float*)d_in[22];
  const float* w_ne = (const float*)d_in[23]; const float* b_ne = (const float*)d_in[24];
  const float* w_c1 = (const float*)d_in[25]; const float* b_c1 = (const float*)d_in[26];
  const float* w_c2 = (const float*)d_in[27]; const float* b_c2 = (const float*)d_in[28];

  // workspace carve (bytes, 256-aligned chunks)
  char* base = (char*)d_ws;
  size_t off = 0;
  auto carve = [&](size_t bytes) {
    void* p = base + off;
    off += (bytes + 255) & ~(size_t)255;
    return p;
  };
  float* node_scores = (float*)carve(N_NODES * 4);
  float* edge_scores = (float*)carve(N_EDGES * 4);
  float* kb_node     = (float*)carve(N_NODES * 4);
  float* xg_n  = (float*)carve((size_t)N_NODES * DMODEL * 4);
  float* xg_e  = (float*)carve((size_t)N_EDGES * DMODEL * 4);
  __bf16* q_n  = (__bf16*)carve((size_t)N_NODES * DMODEL * 2);
  __bf16* k_n  = (__bf16*)carve((size_t)N_NODES * DMODEL * 2);
  __bf16* vt_n = (__bf16*)carve((size_t)DMODEL * N_NODES * 2);
  float* ao_n  = (float*)carve((size_t)N_NODES * DMODEL * 4);
  float* n_out = (float*)carve((size_t)N_NODES * DMODEL * 4);
  __bf16* q_e  = (__bf16*)carve((size_t)N_EDGES * DMODEL * 2);
  __bf16* k_e  = (__bf16*)carve((size_t)N_EDGES * DMODEL * 2);
  __bf16* vt_e = (__bf16*)carve((size_t)DMODEL * N_EDGES * 2);
  float* ao_e  = (float*)carve((size_t)N_EDGES * DMODEL * 4);
  float* e_out = (float*)carve((size_t)N_EDGES * DMODEL * 4);
  float* msg   = (float*)carve((size_t)N_EDGES * DMODEL * 4);
  float* upd   = (float*)carve((size_t)N_EDGES * DMODEL * 4);
  float* hbuf  = (float*)carve((size_t)N_EDGES * 2 * DMODEL * 4);

  // 1) routers (scores + gated features)
  router_kernel<<<(N_NODES + 255) / 256, 256, 0, stream>>>(node_f, w_rn, b_rn,
                                                           node_scores, xg_n, N_NODES);
  router_kernel<<<(N_EDGES + 255) / 256, 256, 0, stream>>>(edge_f, w_re, b_re,
                                                           edge_scores, xg_e, N_EDGES);
  // 2) top-k threshold -> node key bias
  topk_bias_kernel<<<1, 1024, 0, stream>>>(node_scores, kb_node);

  // 3) node QKV projections: Q,K bf16 row-major; V bf16 transposed
  {
    dim3 g(N_NODES / 16, DMODEL / 16);
    gemm_kernel<<<g, 32, 0, stream>>>(xg_n, wq_n, bq_n, nullptr, nullptr, q_n,
                                      N_NODES, DMODEL, DMODEL, 0, 1);
    gemm_kernel<<<g, 32, 0, stream>>>(xg_n, wk_n, bk_n, nullptr, nullptr, k_n,
                                      N_NODES, DMODEL, DMODEL, 0, 1);
    gemm_kernel<<<g, 32, 0, stream>>>(xg_n, wv_n, bv_n, nullptr, nullptr, vt_n,
                                      N_NODES, DMODEL, DMODEL, 0, 2);
  }
  // 4) node attention (top-k key mask)
  attn_kernel<0><<<dim3(N_NODES / 16, NHEAD), 32, 0, stream>>>(
      q_n, k_n, vt_n, kb_node, nullptr, nullptr, ao_n, N_NODES);
  // 5) node output projection (f32)
  gemm_kernel<<<dim3(N_NODES / 16, DMODEL / 16), 32, 0, stream>>>(
      ao_n, wo_n, bo_n, nullptr, n_out, nullptr, N_NODES, DMODEL, DMODEL, 0, 0);

  // 6) edge QKV projections
  {
    dim3 g(N_EDGES / 16, DMODEL / 16);
    gemm_kernel<<<g, 32, 0, stream>>>(xg_e, wq_e, bq_e, nullptr, nullptr, q_e,
                                      N_EDGES, DMODEL, DMODEL, 0, 1);
    gemm_kernel<<<g, 32, 0, stream>>>(xg_e, wk_e, bk_e, nullptr, nullptr, k_e,
                                      N_EDGES, DMODEL, DMODEL, 0, 1);
    gemm_kernel<<<g, 32, 0, stream>>>(xg_e, wv_e, bv_e, nullptr, nullptr, vt_e,
                                      N_EDGES, DMODEL, DMODEL, 0, 2);
  }
  // 7) edge attention (line-graph adjacency mask computed on the fly)
  attn_kernel<1><<<dim3(N_EDGES / 16, NHEAD), 32, 0, stream>>>(
      q_e, k_e, vt_e, nullptr, src, dst, ao_e, N_EDGES);
  // 8) edge output projection (f32)
  gemm_kernel<<<dim3(N_EDGES / 16, DMODEL / 16), 32, 0, stream>>>(
      ao_e, wo_e, bo_e, nullptr, e_out, nullptr, N_EDGES, DMODEL, DMODEL, 0, 0);

  // 9) node message gather: node_out[src] + node_out[dst]
  node_msg_kernel<<<(N_EDGES * 32) / 256, 256, 0, stream>>>(n_out, src, dst, msg);

  // 10) updated_edges = edge_out + msg @ w_ne + b_ne   (residual fused)
  gemm_kernel<<<dim3(N_EDGES / 16, DMODEL / 16), 32, 0, stream>>>(
      msg, w_ne, b_ne, e_out, upd, nullptr, N_EDGES, DMODEL, DMODEL, 0, 0);

  // 11) classifier: gelu(upd @ w_c1 + b_c1) @ w_c2 + b_c2
  gemm_kernel<<<dim3(N_EDGES / 16, (2 * DMODEL) / 16), 32, 0, stream>>>(
      upd, w_c1, b_c1, nullptr, hbuf, nullptr, N_EDGES, DMODEL, 2 * DMODEL, 1, 0);
  gemm_kernel<<<dim3(N_EDGES / 16, NCLS / 16), 32, 0, stream>>>(
      hbuf, w_c2, b_c2, nullptr, (float*)d_out, nullptr, N_EDGES, 2 * DMODEL, NCLS, 0, 0);
}
